// PointOutlierPooling_28372553957670
// MI455X (gfx1250) — compile-verified, matrix operand order checked by reading, not run on hardware
//
#include <hip/hip_runtime.h>
#include <hip/hip_bf16.h>

// ---------------------------------------------------------------- constants
#define B_SZ     16
#define N_PTS    32768
#define IN_C     128
#define X_PAD    160          // 131 padded to K-tiles of 32
#define NUM_OUT  3276         // int(32768 * 0.1)
#define NCLEAN   (N_PTS - NUM_OUT)      // 29492
#define DEC_TILES ((NCLEAN + 63) / 64)  // 461

// workspace layout (bytes)
#define OFF_W1S   0            // 16nt*5kt*512 bf16  = 81920
#define OFF_W2S   81920        // 16*8*512 bf16      = 131072
#define OFF_W3S   212992       // 2*8*512 bf16       = 16384
#define OFF_WD1S  229376       // 4*4*512 bf16       = 16384
#define OFF_WD2S  245760       // 2*2*512 bf16       = 4096
#define OFF_WPN   249856       // float
#define OFF_PROBS 262144       // B*N float          = 2097152
#define OFF_S     2359296      // B*N u64            = 4194304

typedef __bf16 bf16_t;
typedef __attribute__((ext_vector_type(4)))  __bf16 v4bf;
typedef __attribute__((ext_vector_type(8)))  __bf16 v8bf;
typedef __attribute__((ext_vector_type(16))) __bf16 v16bf;
typedef __attribute__((ext_vector_type(8)))  float  v8f;
typedef unsigned long long u64;

// ------------------------------------------------- WMMA fragment utilities
// A-fragment (16x32 bf16): lane holds row m = lane&15.
// element j -> k = 16*(j>=8) + 8*(lane>=16) + (j&7)
__device__ __forceinline__ v16bf load_a_frag(const bf16_t* A, int lda, int m,
                                             int kbase, int lane) {
    const bf16_t* p = A + (size_t)m * lda + kbase + ((lane >> 4) << 3);
    v8bf lo = *(const v8bf*)p;          // k = kbase + 8*(lane>=16) + [0..7]
    v8bf hi = *(const v8bf*)(p + 16);   // k = +16
    return __builtin_shufflevector(lo, hi, 0,1,2,3,4,5,6,7,8,9,10,11,12,13,14,15);
}

// One wave computes a 16-row x (NT*16)-col layer with WMMA, bf16 in, f32 acc.
// ACT: 0 = leaky_relu(0.01), 1 = relu.  TO_BF16 selects output type statically.
template <int ACT, int KT, int NT, bool TO_BF16>
__device__ __forceinline__ void wave_mlp_layer(
    const bf16_t* __restrict__ A, int lda, int row0,
    const bf16_t* __restrict__ Wfrag,
    const float* __restrict__ bias,
    void* __restrict__ Out, int ldo, int lane)
{
    const int m     = row0 + (lane & 15);
    const int ncol  = lane & 15;
    const int rhalf = (lane >> 4) << 3;

    // A is invariant across column tiles: hoist all fragments into registers.
    v16bf afr[KT];
    #pragma unroll
    for (int kt = 0; kt < KT; ++kt)
        afr[kt] = load_a_frag(A, lda, m, kt * 32, lane);

    // Preload all per-tile bias values (removes dependent load from hot loop).
    float bvs[NT];
    #pragma unroll
    for (int nt = 0; nt < NT; ++nt) bvs[nt] = bias[nt * 16 + ncol];

    for (int nt = 0; nt < NT; ++nt) {
        const v16bf* wbase = ((const v16bf*)Wfrag) + (size_t)nt * KT * 32 + lane;
        __builtin_prefetch((const void*)(wbase + (size_t)KT * 32), 0, 3);
        v8f c;
        #pragma unroll
        for (int r = 0; r < 8; ++r) c[r] = bvs[nt];
        #pragma unroll
        for (int kt = 0; kt < KT; ++kt) {
            v16bf b = wbase[kt * 32];
            c = __builtin_amdgcn_wmma_f32_16x16x32_bf16(
                    false, afr[kt], false, b, (short)0, c, false, false);
        }
        #pragma unroll
        for (int r = 0; r < 8; ++r) {
            float v = c[r];
            v = (ACT == 0) ? (v > 0.f ? v : 0.01f * v) : (v > 0.f ? v : 0.f);
            size_t o = (size_t)(row0 + rhalf + r) * ldo + nt * 16 + ncol;
            if (TO_BF16) ((bf16_t*)Out)[o] = (bf16_t)v;
            else         ((float*)Out)[o]  = v;
        }
    }
}

// ------------------------------------------------------- weight swizzling
// Fragment layout: [nt][kt][lane][j] (each lane's 16 bf16 contiguous = 32B).
__global__ __launch_bounds__(256) void swizzle_weights_kernel(
    const float* __restrict__ W1, const float* __restrict__ W2,
    const float* __restrict__ W3, const float* __restrict__ Wd1,
    const float* __restrict__ Wd2, const float* __restrict__ Wp,
    bf16_t* W1s, bf16_t* W2s, bf16_t* W3s, bf16_t* Wd1s, bf16_t* Wd2s,
    float* wpnorm)
{
    const int n1 = 16*5*512, n2 = 16*8*512, n3 = 2*8*512, n4 = 4*4*512, n5 = 2*2*512;
    const int total = n1 + n2 + n3 + n4 + n5;
    int gid = blockIdx.x * 256 + threadIdx.x;
    const float* src; bf16_t* dst; int KT, Ksrc, Ncols; int e;
    if      (gid < n1)           { e = gid;              src = W1;  dst = W1s;  KT = 5; Ksrc = 131; Ncols = 256; }
    else if (gid < n1+n2)        { e = gid-n1;           src = W2;  dst = W2s;  KT = 8; Ksrc = 256; Ncols = 256; }
    else if (gid < n1+n2+n3)     { e = gid-n1-n2;        src = W3;  dst = W3s;  KT = 8; Ksrc = 256; Ncols = 32;  }
    else if (gid < n1+n2+n3+n4)  { e = gid-n1-n2-n3;     src = Wd1; dst = Wd1s; KT = 4; Ksrc = 128; Ncols = 64;  }
    else if (gid < total)        { e = gid-n1-n2-n3-n4;  src = Wd2; dst = Wd2s; KT = 2; Ksrc = 64;  Ncols = 32;  }
    else {
        if (gid == total) {
            float s = 0.f;
            for (int i = 0; i < 32; ++i) s += Wp[i] * Wp[i];
            *wpnorm = sqrtf(s);
        }
        return;
    }
    int j = e & 15, lane = (e >> 4) & 31, fid = e >> 9;
    int kt = fid % KT, nt = fid / KT;
    int k = kt * 32 + ((j >> 3) << 4) + ((lane >> 4) << 3) + (j & 7);
    int n = nt * 16 + (lane & 15);
    float v = (k < Ksrc) ? src[k * Ncols + n] : 0.f;
    dst[e] = (bf16_t)v;
}

// ---------------------------------------------------------- encoder (MLP)
__global__ __launch_bounds__(128) void encoder_kernel(
    const float* __restrict__ xyz, const float* __restrict__ f,
    const bf16_t* __restrict__ W1s, const float* __restrict__ b1,
    const bf16_t* __restrict__ W2s, const float* __restrict__ b2,
    const bf16_t* __restrict__ W3s, const float* __restrict__ b3,
    const float* __restrict__ Wp, const float* __restrict__ bp,
    const float* __restrict__ wpnorm,
    float* __restrict__ probs, u64* __restrict__ S)
{
    // 64KB LDS, overlaid buffers: xs[64*160]bf16 | h1[64*256]bf16 | h2 | h3 f32
    __shared__ __align__(16) char smem[65536];
    bf16_t* xs = (bf16_t*)smem;              // [0, 20480)
    bf16_t* h1 = (bf16_t*)(smem + 32768);    // [32768, 65536)
    bf16_t* h2 = (bf16_t*)smem;             // [0, 32768)  (xs dead)
    float*  h3 = (float*)(smem + 32768);    // [32768, 40960) (h1 dead)

    int tid = threadIdx.x, wave = tid >> 5, lane = tid & 31;
    int b   = blockIdx.x / (N_PTS / 64);
    int pt0 = (blockIdx.x % (N_PTS / 64)) * 64;
    size_t rowbase = (size_t)b * N_PTS + pt0;

    // stage f: float4 loads, packed v4bf LDS stores
    for (int i = tid; i < 64 * 32; i += 128) {
        int r = i >> 5, c4 = i & 31;
        float4 v = ((const float4*)(f + (rowbase + r) * IN_C))[c4];
        v4bf o;
        o[0] = (bf16_t)v.x; o[1] = (bf16_t)v.y;
        o[2] = (bf16_t)v.z; o[3] = (bf16_t)v.w;
        *(v4bf*)&xs[r * X_PAD + c4 * 4] = o;
    }
    for (int i = tid; i < 64 * 32; i += 128) {
        int r = i >> 5, c = i & 31;
        float v = (c < 3) ? xyz[(rowbase + r) * 3 + c] : 0.f;
        xs[r * X_PAD + 128 + c] = (bf16_t)v;
    }
    __syncthreads();

    int row0 = wave * 16;
    wave_mlp_layer<0, 5, 16, true >(xs, X_PAD, row0, W1s, b1, h1, 256, lane);
    __syncthreads();
    wave_mlp_layer<0, 8, 16, true >(h1, 256, row0, W2s, b2, h2, 256, lane);
    __syncthreads();
    wave_mlp_layer<1, 8, 2, false>(h2, 256, row0, W3s, b3, h3, 32, lane);
    __syncthreads();

    if (lane < 16) {
        int r = row0 + lane;
        float acc = bp[0];
        #pragma unroll
        for (int c = 0; c < 32; ++c) acc += h3[r * 32 + c] * Wp[c];
        float p = acc / wpnorm[0];
        size_t gi = rowbase + r;
        probs[gi] = p;
        unsigned int bits = __float_as_uint(p);
        unsigned int key  = bits ^ ((bits >> 31) ? 0xFFFFFFFFu : 0x80000000u);
        key = ~key;  // ascending u64 sort => descending prob, stable (idx asc)
        S[gi] = ((u64)key << 32) | (unsigned int)(pt0 + r);
    }
}

// ------------------------------------------------------ bitonic sort (u64)
#define CHUNK 4096
__global__ __launch_bounds__(1024) void bitonic_local_kernel(
    u64* __restrict__ S, unsigned kstart, unsigned kend)
{
    __shared__ u64 ls[CHUNK];   // 32KB
    int tid = threadIdx.x;
    size_t base = (size_t)blockIdx.x * CHUNK;  // N multiple of CHUNK: no straddle
    for (int i = tid; i < CHUNK; i += 1024) ls[i] = S[base + i];
    __syncthreads();
    for (unsigned k = kstart; k <= kend; k <<= 1) {
        unsigned j0 = ((k >> 1) < (CHUNK >> 1)) ? (k >> 1) : (CHUNK >> 1);
        for (unsigned j = j0; j > 0; j >>= 1) {
            for (int i = tid; i < CHUNK; i += 1024) {
                unsigned p = (unsigned)i ^ j;
                if (p > (unsigned)i) {
                    unsigned li = (unsigned)((base + i) & (N_PTS - 1));
                    bool up = ((li & k) == 0);
                    u64 a = ls[i], c = ls[p];
                    if (up ? (a > c) : (a < c)) { ls[i] = c; ls[p] = a; }
                }
            }
            __syncthreads();
        }
    }
    for (int i = tid; i < CHUNK; i += 1024) S[base + i] = ls[i];
}

__global__ __launch_bounds__(256) void bitonic_global_kernel(
    u64* __restrict__ S, unsigned k, unsigned j)
{
    unsigned t = blockIdx.x * 256 + threadIdx.x;     // B*N/2 threads
    unsigned i = (t / j) * (2u * j) + (t % j);
    unsigned p = i + j;
    unsigned li = i & (N_PTS - 1);
    bool up = ((li & k) == 0);
    u64 a = S[i], c = S[p];
    if (up ? (a > c) : (a < c)) { S[i] = c; S[p] = a; }
}

__global__ __launch_bounds__(256) void emit_idx_kernel(
    const u64* __restrict__ S, int* __restrict__ out)
{
    size_t gi = (size_t)blockIdx.x * 256 + threadIdx.x;
    out[gi] = (int)(unsigned)(S[gi] & 0xFFFFFFFFull);
}

// ---------------------------------------------------------------- decoder
__global__ __launch_bounds__(128) void decoder_kernel(
    const float* __restrict__ xyz, const float* __restrict__ f,
    const bf16_t* __restrict__ Wd1s, const float* __restrict__ bd1,
    const bf16_t* __restrict__ Wd2s, const float* __restrict__ bd2,
    const float* __restrict__ Wd3, const float* __restrict__ bd3,
    const float* __restrict__ probs, const u64* __restrict__ S,
    float* __restrict__ est)
{
    __shared__ __align__(16) bf16_t ps[64 * 128];  // 16KB
    __shared__ __align__(16) bf16_t hh[64 * 64];   // 8KB
    __shared__ float gg[64 * 32];                  // 8KB
    __shared__ int   cidx[64];
    __shared__ float yv[64];

    int tid = threadIdx.x, wave = tid >> 5, lane = tid & 31;
    int b  = blockIdx.x / DEC_TILES;
    int t0 = (blockIdx.x % DEC_TILES) * 64;
    int rows = NCLEAN - t0; if (rows > 64) rows = 64;

    if (tid < 64) {
        int idx = 0; float y = 0.f;
        if (tid < rows) {
            size_t pos = (size_t)b * N_PTS + NUM_OUT + t0 + tid;
            idx = (int)(unsigned)(S[pos] & 0xFFFFFFFFull);
            float p = probs[(size_t)b * N_PTS + idx];
            y = 1.f / (1.f + __expf(-p));
        }
        cidx[tid] = idx; yv[tid] = y;
    }
    __syncthreads();
    for (int i = tid; i < 64 * 32; i += 128) {
        int r = i >> 5, c4 = i & 31;
        float4 v = ((const float4*)(f + ((size_t)b * N_PTS + cidx[r]) * IN_C))[c4];
        float y = yv[r];
        v4bf o;
        o[0] = (bf16_t)(v.x * y); o[1] = (bf16_t)(v.y * y);
        o[2] = (bf16_t)(v.z * y); o[3] = (bf16_t)(v.w * y);
        *(v4bf*)&ps[r * 128 + c4 * 4] = o;
    }
    __syncthreads();

    int row0 = wave * 16;
    wave_mlp_layer<1, 4, 4, true >(ps, 128, row0, Wd1s, bd1, hh, 64, lane);
    __syncthreads();
    wave_mlp_layer<1, 2, 2, false>(hh, 64, row0, Wd2s, bd2, gg, 32, lane);
    __syncthreads();

    if (tid < 64 && tid < rows) {
        int r = tid;
        float d0 = bd3[0], d1 = bd3[1], d2v = bd3[2];
        #pragma unroll
        for (int c = 0; c < 32; ++c) {
            float g = gg[r * 32 + c];
            d0 += g * Wd3[c * 3 + 0]; d1 += g * Wd3[c * 3 + 1]; d2v += g * Wd3[c * 3 + 2];
        }
        size_t src = ((size_t)b * N_PTS + cidx[r]) * 3;
        size_t dst = ((size_t)b * NCLEAN + t0 + r) * 3;
        est[dst + 0] = xyz[src + 0] + d0;
        est[dst + 1] = xyz[src + 1] + d1;
        est[dst + 2] = xyz[src + 2] + d2v;
    }
}

// ------------------------------------------------------------------ launch
extern "C" void kernel_launch(void* const* d_in, const int* in_sizes, int n_in,
                              void* d_out, int out_size, void* d_ws, size_t ws_size,
                              hipStream_t stream)
{
    const float* xyz = (const float*)d_in[0];
    const float* f   = (const float*)d_in[1];
    const float* W1  = (const float*)d_in[2];
    const float* b1  = (const float*)d_in[3];
    const float* W2  = (const float*)d_in[4];
    const float* b2  = (const float*)d_in[5];
    const float* W3  = (const float*)d_in[6];
    const float* b3  = (const float*)d_in[7];
    const float* Wp  = (const float*)d_in[8];
    const float* bp  = (const float*)d_in[9];
    const float* Wd1 = (const float*)d_in[10];
    const float* bd1 = (const float*)d_in[11];
    const float* Wd2 = (const float*)d_in[12];
    const float* bd2 = (const float*)d_in[13];
    const float* Wd3 = (const float*)d_in[14];
    const float* bd3 = (const float*)d_in[15];

    char* ws = (char*)d_ws;
    bf16_t* W1s  = (bf16_t*)(ws + OFF_W1S);
    bf16_t* W2s  = (bf16_t*)(ws + OFF_W2S);
    bf16_t* W3s  = (bf16_t*)(ws + OFF_W3S);
    bf16_t* Wd1s = (bf16_t*)(ws + OFF_WD1S);
    bf16_t* Wd2s = (bf16_t*)(ws + OFF_WD2S);
    float*  wpn  = (float*)(ws + OFF_WPN);
    float*  probs= (float*)(ws + OFF_PROBS);
    u64*    S    = (u64*)(ws + OFF_S);

    int*   out_idx = (int*)d_out;
    float* out_est = ((float*)d_out) + (size_t)B_SZ * N_PTS;

    swizzle_weights_kernel<<<489, 256, 0, stream>>>(
        W1, W2, W3, Wd1, Wd2, Wp, W1s, W2s, W3s, Wd1s, Wd2s, wpn);

    encoder_kernel<<<(B_SZ * N_PTS) / 64, 128, 0, stream>>>(
        xyz, f, W1s, b1, W2s, b2, W3s, b3, Wp, bp, wpn, probs, S);

    const int LGRID = (B_SZ * N_PTS) / CHUNK;           // 128
    const int GGRID = (B_SZ * N_PTS / 2) / 256;         // 1024
    bitonic_local_kernel<<<LGRID, 1024, 0, stream>>>(S, 2u, (unsigned)CHUNK);
    bitonic_global_kernel<<<GGRID, 256, 0, stream>>>(S, 8192u, 4096u);
    bitonic_local_kernel<<<LGRID, 1024, 0, stream>>>(S, 8192u, 8192u);
    bitonic_global_kernel<<<GGRID, 256, 0, stream>>>(S, 16384u, 8192u);
    bitonic_global_kernel<<<GGRID, 256, 0, stream>>>(S, 16384u, 4096u);
    bitonic_local_kernel<<<LGRID, 1024, 0, stream>>>(S, 16384u, 16384u);
    bitonic_global_kernel<<<GGRID, 256, 0, stream>>>(S, 32768u, 16384u);
    bitonic_global_kernel<<<GGRID, 256, 0, stream>>>(S, 32768u, 8192u);
    bitonic_global_kernel<<<GGRID, 256, 0, stream>>>(S, 32768u, 4096u);
    bitonic_local_kernel<<<LGRID, 1024, 0, stream>>>(S, 32768u, 32768u);

    emit_idx_kernel<<<(B_SZ * N_PTS) / 256, 256, 0, stream>>>(S, out_idx);

    decoder_kernel<<<B_SZ * DEC_TILES, 128, 0, stream>>>(
        xyz, f, Wd1s, bd1, Wd2s, bd2, Wd3, bd3, probs, S, out_est);
}
